// Model12_64630667870281
// MI455X (gfx1250) — matrix operand order repaired and checked by reading, not compiled, not run on hardware
//
#include <hip/hip_runtime.h>
#include <hip/hip_bf16.h>

typedef __bf16 bf16_t;
typedef __attribute__((ext_vector_type(16))) __bf16 v16bf;
typedef __attribute__((ext_vector_type(8)))  float  v8f;
typedef __attribute__((ext_vector_type(4)))  float  v4f;

#define N_NODES_C  100000
#define N_EDGES_C  1600000
#define D_FEAT_C   64
#define N_GRAPHS_C 1000
#define D_MSG_C    30
#define NTILES_C   (N_EDGES_C / 16)   // 100000 tiles of 16 edges

// ---------------------------------------------------------------------------
// Kernel 0: zero the accumulator workspace (harness poisons ws once with 0xAA
// and never re-poisons between replays; we must re-zero every call).
// ---------------------------------------------------------------------------
__global__ void zero_ws_kernel(float* __restrict__ p, long n) {
  long i = (long)blockIdx.x * blockDim.x + threadIdx.x;
  long stride = (long)gridDim.x * blockDim.x;
  for (; i < n; i += stride) p[i] = 0.0f;
}

// Which source array does feature-column K of the concat [x_src | x_dst | e]
// come from?  K is always a multiple of 8 here, so an 8-float run never
// crosses a segment boundary.  All offsets fit in 32 bits.
__device__ __forceinline__ const float* seg_ptr(int K,
                                                const float* __restrict__ na,
                                                const float* __restrict__ ea,
                                                int s, int d, int e) {
  if (K < 64)  return na + (unsigned)(s * D_FEAT_C + K);
  if (K < 128) return na + (unsigned)(d * D_FEAT_C + (K - 64));
  return ea + (unsigned)(e * D_FEAT_C + (K - 128));
}

// ---------------------------------------------------------------------------
// Kernel 1: per-edge message GEMM [16 edges x 192] x [192 x 30] via bf16 WMMA,
// ReLU, atomic scatter-add into x_nodes[dst].
// One wave32 per 16-edge tile; W_msg fragments register-resident per wave.
// __launch_bounds__(256,1) relaxes the VGPR budget (~160 live VGPRs -> still
// 4 waves/SIMD) so the 12 B fragments do NOT spill to scratch.
// ---------------------------------------------------------------------------
__global__ void __launch_bounds__(256, 1)
edge_msg_kernel(const int*   __restrict__ edge_index,   // [2*E]
                const float* __restrict__ node_attr,    // [N,64]
                const float* __restrict__ edge_attr,    // [E,64]
                const float* __restrict__ W_msg,        // [192,30]
                const float* __restrict__ b_msg,        // [30]
                float*       __restrict__ x_nodes) {    // [N,30]
  const int lane   = threadIdx.x & 31;
  const int waveId = threadIdx.x >> 5;
  const int wavesPerBlock = blockDim.x >> 5;
  const int half = lane >> 4;   // 0: lanes 0-15, 1: lanes 16-31
  const int r    = lane & 15;

  // --- B fragments: W_msg (192x30 f32) -> bf16, padded to 32 cols.
  // B layout (32x16 16-bit): lane holds column n = lane&15; element j <-> K =
  // (j&7) + 8*half + 16*(j>>3) (+ 32*kstep).  Two N-tiles cover cols 0..15 and 16..31.
  v16bf bfrag[2][6];
  #pragma unroll
  for (int t = 0; t < 2; ++t) {
    const int n = r + 16 * t;
    #pragma unroll
    for (int ks = 0; ks < 6; ++ks) {
      #pragma unroll
      for (int j = 0; j < 16; ++j) {
        const int K = (j & 7) + 8 * half + 16 * (j >> 3) + 32 * ks;
        const float w = (n < D_MSG_C) ? W_msg[K * D_MSG_C + n] : 0.0f;
        bfrag[t][ks][j] = (bf16_t)w;
      }
    }
  }
  const float bias0 = b_msg[r];                                  // col r
  const float bias1 = (r < D_MSG_C - 16) ? b_msg[r + 16] : 0.0f; // col r+16

  for (int tile = blockIdx.x * wavesPerBlock + waveId; tile < NTILES_C;
       tile += gridDim.x * wavesPerBlock) {
    const int tileBase = tile * 16;
    const int eRow = tileBase + r;                 // this lane's A-row edge
    const int s = edge_index[eRow];                // src node
    const int d = edge_index[N_EDGES_C + eRow];    // dst node

    v8f c0 = {};  // cols 0..15
    v8f c1 = {};  // cols 16..29 (30,31 are zero-padded weights)

    #pragma unroll
    for (int ks = 0; ks < 6; ++ks) {
      // A layout (16x32 16-bit): element j <-> K = j + 8*half (j<8),
      //                                       K = j + 8 + 8*half (j>=8), +32*ks.
      const int Kb = 32 * ks + 8 * half;
      const float* p = seg_ptr(Kb, node_attr, edge_attr, s, d, eRow);
      const v4f f0 = *(const v4f*)(p);        // K = Kb   .. Kb+3
      const v4f f1 = *(const v4f*)(p + 4);    // K = Kb+4 .. Kb+7
      const v4f f2 = *(const v4f*)(p + 16);   // K = Kb+16.. Kb+19
      const v4f f3 = *(const v4f*)(p + 20);   // K = Kb+20.. Kb+23
      v16bf a;
      #pragma unroll
      for (int q = 0; q < 4; ++q) {
        a[q]      = (bf16_t)f0[q];
        a[4 + q]  = (bf16_t)f1[q];
        a[8 + q]  = (bf16_t)f2[q];
        a[12 + q] = (bf16_t)f3[q];
      }
      c0 = __builtin_amdgcn_wmma_f32_16x16x32_bf16(false, a, false, bfrag[0][ks],
                                                   (short)0, c0, false, false);
      c1 = __builtin_amdgcn_wmma_f32_16x16x32_bf16(false, a, false, bfrag[1][ks],
                                                   (short)0, c1, false, false);
    }

    // C/D layout: VGPR v, lane -> row m = v + 8*half, col n = r (+16 for c1).
    #pragma unroll
    for (int v = 0; v < 8; ++v) {
      const int e  = tileBase + v + 8 * half;
      const int dv = edge_index[N_EDGES_C + e];   // dst of row-m edge (L1-hot)
      const float val0 = fmaxf(c0[v] + bias0, 0.0f);
      atomicAdd(&x_nodes[(unsigned)(dv * D_MSG_C + r)], val0);
      if (r < D_MSG_C - 16) {
        const float val1 = fmaxf(c1[v] + bias1, 0.0f);
        atomicAdd(&x_nodes[(unsigned)(dv * D_MSG_C + r + 16)], val1);
      }
    }
  }
}

// ---------------------------------------------------------------------------
// Kernel 2: per-node MLP head relu(x @ W1 + b1) [30->20] + atomic graph pool.
// ---------------------------------------------------------------------------
__global__ void node_mlp_kernel(const float* __restrict__ x_nodes,  // [N,30]
                                const float* __restrict__ W1,       // [30,20]
                                const float* __restrict__ b1,       // [20]
                                const int*   __restrict__ batch,    // [N]
                                float*       __restrict__ g) {      // [G,20]
  __shared__ float sW[D_MSG_C * 20];
  __shared__ float sb[20];
  for (int i = threadIdx.x; i < D_MSG_C * 20; i += blockDim.x) sW[i] = W1[i];
  if (threadIdx.x < 20) sb[threadIdx.x] = b1[threadIdx.x];
  __syncthreads();

  const int i = blockIdx.x * blockDim.x + threadIdx.x;
  if (i >= N_NODES_C) return;

  float h[20];
  #pragma unroll
  for (int j = 0; j < 20; ++j) h[j] = sb[j];

  const float* xr = x_nodes + (unsigned)(i * D_MSG_C);
  for (int k = 0; k < D_MSG_C; ++k) {
    const float xv = xr[k];
    #pragma unroll
    for (int j = 0; j < 20; ++j) h[j] = fmaf(xv, sW[k * 20 + j], h[j]);
  }

  const int gi = batch[i];
  #pragma unroll
  for (int j = 0; j < 20; ++j)
    atomicAdd(&g[(unsigned)(gi * 20 + j)], fmaxf(h[j], 0.0f));
}

// ---------------------------------------------------------------------------
// Kernel 3: per-graph head relu(g @ W2 + b2) [20->10] @ W3 + b3 -> out [G,1].
// ---------------------------------------------------------------------------
__global__ void graph_head_kernel(const float* __restrict__ g,    // [G,20]
                                  const float* __restrict__ W2,   // [20,10]
                                  const float* __restrict__ b2,   // [10]
                                  const float* __restrict__ W3,   // [10,1]
                                  const float* __restrict__ b3,   // [1]
                                  float*       __restrict__ out) {// [G]
  const int i = blockIdx.x * blockDim.x + threadIdx.x;
  if (i >= N_GRAPHS_C) return;

  float h[10];
  #pragma unroll
  for (int j = 0; j < 10; ++j) h[j] = b2[j];
  const float* gr = g + (unsigned)(i * 20);
  for (int k = 0; k < 20; ++k) {
    const float v = gr[k];
    #pragma unroll
    for (int j = 0; j < 10; ++j) h[j] = fmaf(v, W2[k * 10 + j], h[j]);
  }
  float o = b3[0];
  #pragma unroll
  for (int j = 0; j < 10; ++j) o = fmaf(fmaxf(h[j], 0.0f), W3[j], o);
  out[i] = o;
}

// ---------------------------------------------------------------------------
extern "C" void kernel_launch(void* const* d_in, const int* in_sizes, int n_in,
                              void* d_out, int out_size, void* d_ws, size_t ws_size,
                              hipStream_t stream) {
  const int*   edge_index = (const int*)  d_in[0];   // [2, E]
  const float* node_attr  = (const float*)d_in[1];   // [N, 64]
  const float* edge_attr  = (const float*)d_in[2];   // [E, 64]
  const int*   batch      = (const int*)  d_in[3];   // [N]
  const float* W_msg      = (const float*)d_in[4];   // [192, 30]
  const float* b_msg      = (const float*)d_in[5];   // [30]
  const float* W1         = (const float*)d_in[6];   // [30, 20]
  const float* b1         = (const float*)d_in[7];   // [20]
  const float* W2         = (const float*)d_in[8];   // [20, 10]
  const float* b2         = (const float*)d_in[9];   // [10]
  const float* W3         = (const float*)d_in[10];  // [10, 1]
  const float* b3         = (const float*)d_in[11];  // [1]
  float* out = (float*)d_out;

  float* x_nodes = (float*)d_ws;                           // [N, 30]
  float* g_pool  = x_nodes + (long)N_NODES_C * D_MSG_C;    // [G, 20]
  const long zero_n = (long)N_NODES_C * D_MSG_C + (long)N_GRAPHS_C * 20;

  zero_ws_kernel<<<1024, 256, 0, stream>>>(x_nodes, zero_n);

  // 100000 tiles; 2048 blocks x 8 waves -> ~6 tiles/wave (amortizes B-fragment load)
  edge_msg_kernel<<<2048, 256, 0, stream>>>(edge_index, node_attr, edge_attr,
                                            W_msg, b_msg, x_nodes);

  node_mlp_kernel<<<(N_NODES_C + 255) / 256, 256, 0, stream>>>(x_nodes, W1, b1,
                                                               batch, g_pool);

  graph_head_kernel<<<(N_GRAPHS_C + 255) / 256, 256, 0, stream>>>(g_pool, W2, b2,
                                                                  W3, b3, out);
}
